// CombinedLoss_77369540870626
// MI455X (gfx1250) — matrix operand order repaired and checked by reading, compiled, not run on hardware
//
#include <hip/hip_runtime.h>
#include <math.h>

typedef __attribute__((ext_vector_type(16))) __bf16 v16bf;
typedef __attribute__((ext_vector_type(8)))  __bf16 v8bf;
typedef __attribute__((ext_vector_type(8)))  float  v8f;

#define BDIM 32
#define PDIM 1024
#define HDIM 768
#define NROW (BDIM * PDIM)
#define NGEMMBLK (BDIM * 64)   // 64 row-tiles of 16 cover P=1024
#define IGNORE_L (-100)
#define INV_TEMP (1.0f / 0.07f)
#define NEG_FILL (-1e9f)

// ---- bf16 helpers: manual RNE convert ----
__device__ __forceinline__ __bf16 f2bf(float x) {
  unsigned u = __builtin_bit_cast(unsigned, x);
  unsigned r = (u + 0x7FFFu + ((u >> 16) & 1u)) >> 16;
  return __builtin_bit_cast(__bf16, (unsigned short)r);
}
__device__ __forceinline__ float bf2f(__bf16 h) {
  unsigned u = ((unsigned)__builtin_bit_cast(unsigned short, h)) << 16;
  return __builtin_bit_cast(float, u);
}

// ---- per-lane online logsumexp primitives ----
__device__ __forceinline__ void lse_update(float& m, float& s, float x) {
  float M2 = fmaxf(m, x);
  s = s * expf(m - M2) + expf(x - M2);
  m = M2;
}
__device__ __forceinline__ void lse_merge(float& m, float& s, float om, float os) {
  float M2 = fmaxf(m, om);
  s = s * expf(m - M2) + os * expf(om - M2);
  m = M2;
}

// ---------------------------------------------------------------------------
// Pass 1: per-row norms, diagonal cosine/TEMP, per-row CE. One wave per row.
// ---------------------------------------------------------------------------
__global__ void prep_kernel(const float* __restrict__ logits,
                            const int*   __restrict__ labels,
                            const float* __restrict__ g,
                            const float* __restrict__ e,
                            float* __restrict__ invg, float* __restrict__ inve,
                            float* __restrict__ diag, float* __restrict__ ce) {
  int wave = (blockIdx.x * blockDim.x + threadIdx.x) >> 5;
  int lane = threadIdx.x & 31;
  if (wave >= NROW) return;
  const float* gr = g + (size_t)wave * HDIM;
  const float* er = e + (size_t)wave * HDIM;
  float sg = 0.f, se = 0.f, d = 0.f;
  for (int i = lane; i < HDIM; i += 32) {
    float gv = gr[i], ev = er[i];
    sg += gv * gv; se += ev * ev; d += gv * ev;
  }
  for (int m = 16; m >= 1; m >>= 1) {
    sg += __shfl_xor(sg, m, 32);
    se += __shfl_xor(se, m, 32);
    d  += __shfl_xor(d,  m, 32);
  }
  if (lane == 0) {
    float ivg = 1.f / fmaxf(sqrtf(sg), 1e-12f);
    float ive = 1.f / fmaxf(sqrtf(se), 1e-12f);
    invg[wave] = ivg;
    inve[wave] = ive;
    diag[wave] = d * ivg * ive * INV_TEMP;          // pos_sim (already /TEMP)
    int lb = labels[wave];
    float l0 = logits[2 * wave], l1 = logits[2 * wave + 1];
    float mx = fmaxf(l0, l1);
    float lse = mx + logf(expf(l0 - mx) + expf(l1 - mx));
    ce[wave] = (lb != IGNORE_L) ? (lse - (lb == 1 ? l1 : l0)) : 0.f;
  }
}

// ---------------------------------------------------------------------------
// Pass 2: deterministic compaction (LDS prefix scan). One block per batch.
// ---------------------------------------------------------------------------
__global__ void compact_kernel(const int* __restrict__ labels,
                               int* __restrict__ pos_list, int* __restrict__ neg_list,
                               int* __restrict__ npos, int* __restrict__ nneg) {
  __shared__ int sp[PDIM];
  __shared__ int sn[PDIM];
  int b = blockIdx.x, p = threadIdx.x;
  int lb = labels[b * PDIM + p];
  int isp = (lb == 1) ? 1 : 0;
  int isn = (lb == 0) ? 1 : 0;
  sp[p] = isp; sn[p] = isn;
  __syncthreads();
  for (int off = 1; off < PDIM; off <<= 1) {
    int vp = 0, vn = 0;
    if (p >= off) { vp = sp[p - off]; vn = sn[p - off]; }
    __syncthreads();
    sp[p] += vp; sn[p] += vn;
    __syncthreads();
  }
  if (isp) pos_list[b * PDIM + sp[p] - 1] = p;
  if (isn) neg_list[b * PDIM + sn[p] - 1] = p;
  if (p == PDIM - 1) { npos[b] = sp[p]; nneg[b] = sn[p]; }
}

// ---------------------------------------------------------------------------
// Pass 2.5 (fast path): convert normalized compacted rows to bf16 hi/lo once.
// Wave w < NROW: positive g-rows -> Gh/Gl; w >= NROW: negative e-rows -> Eh/El.
// ---------------------------------------------------------------------------
__global__ void convert_kernel(const float* __restrict__ g, const float* __restrict__ e,
                               const float* __restrict__ invg, const float* __restrict__ inve,
                               const float* __restrict__ diag,
                               const int* __restrict__ pos_list, const int* __restrict__ neg_list,
                               const int* __restrict__ npos_a, const int* __restrict__ nneg_a,
                               __bf16* __restrict__ Gh, __bf16* __restrict__ Gl,
                               __bf16* __restrict__ Eh, __bf16* __restrict__ El,
                               float* __restrict__ posdiag) {
  int wave = (blockIdx.x * blockDim.x + threadIdx.x) >> 5;
  int lane = threadIdx.x & 31;
  int side = (wave >= NROW) ? 1 : 0;
  int sgl  = side ? (wave - NROW) : wave;
  int b = sgl >> 10, slot = sgl & (PDIM - 1);
  int cnt = side ? nneg_a[b] : npos_a[b];
  if (slot >= cnt) return;
  int idx = (side ? neg_list : pos_list)[b * PDIM + slot];
  const float* src = (side ? e : g) + (size_t)(b * PDIM + idx) * HDIM;
  float scale = (side ? inve : invg)[b * PDIM + idx];
  __bf16* dh = (side ? Eh : Gh) + (size_t)(b * PDIM + slot) * HDIM;
  __bf16* dl = (side ? El : Gl) + (size_t)(b * PDIM + slot) * HDIM;
  for (int i = lane; i < HDIM; i += 32) {
    float x = src[i] * scale;            // normalization folded in
    __bf16 h = f2bf(x);
    dh[i] = h;
    dl[i] = f2bf(x - bf2f(h));
  }
  if (!side && lane == 0) posdiag[b * PDIM + slot] = diag[b * PDIM + idx];
}

// ---------------------------------------------------------------------------
// Pass 3 (fast path): bf16 hi/lo GEMM (x3 WMMA), 2 column chunks per A-load,
// per-lane online logsumexp (no cross-lane ops in the loop).
// One wave per (batch, 16-positive-row tile).
// ---------------------------------------------------------------------------
__global__ void __launch_bounds__(32)
gemm_lse_fast_kernel(const __bf16* __restrict__ Gh, const __bf16* __restrict__ Gl,
                     const __bf16* __restrict__ Eh, const __bf16* __restrict__ El,
                     const float* __restrict__ posdiag,
                     const int* __restrict__ npos_a, const int* __restrict__ nneg_a,
                     float* __restrict__ part) {
  int blk  = blockIdx.x;
  int b    = blk >> 6;
  int tile = blk & 63;
  int lane = threadIdx.x;
  int m15  = lane & 15;
  int npos = npos_a[b], nneg = nneg_a[b];
  bool active = (nneg > 0) && (tile * 16 < npos);
  float final_v = 0.f;

  if (active) {
    int kbaseA = (lane < 16) ? 0 : 16;
    int kbaseB = (lane < 16) ? 0 : 16;
    int half8  = (lane < 16) ? 0 : 8;
    int rslot  = tile * 16 + m15;
    int rsl    = (rslot < npos) ? rslot : (npos - 1);
    const __bf16* ghrow = Gh + (size_t)(b * PDIM + rsl) * HDIM + kbaseA;
    const __bf16* glrow = Gl + (size_t)(b * PDIM + rsl) * HDIM + kbaseA;

    float rdiag[8]; bool rvalid[8];
#pragma unroll
    for (int v = 0; v < 8; ++v) {
      int slot = tile * 16 + v + half8;
      rvalid[v] = (slot < npos);
      rdiag[v]  = posdiag[b * PDIM + (rvalid[v] ? slot : (npos - 1))];
    }

    float rm[8], rs[8];
#pragma unroll
    for (int v = 0; v < 8; ++v) { rm[v] = -1e30f; rs[v] = 0.f; }

    int nchunk = (nneg + 15) >> 4;
    for (int ch = 0; ch < nchunk; ch += 2) {
      int c0 = ch * 16 + m15, c1 = c0 + 16;
      bool v0 = (c0 < nneg);
      bool v1 = (ch + 1 < nchunk) && (c1 < nneg);
      int cs0 = (c0 < nneg) ? c0 : (nneg - 1);
      int cs1 = (c1 < nneg) ? c1 : (nneg - 1);
      const __bf16* e0h = Eh + (size_t)(b * PDIM + cs0) * HDIM + kbaseB;
      const __bf16* e0l = El + (size_t)(b * PDIM + cs0) * HDIM + kbaseB;
      const __bf16* e1h = Eh + (size_t)(b * PDIM + cs1) * HDIM + kbaseB;
      const __bf16* e1l = El + (size_t)(b * PDIM + cs1) * HDIM + kbaseB;

      v8f acc0 = {}, acc1 = {};
      for (int k0 = 0; k0 < HDIM; k0 += 32) {
        v8bf a0 = *(const v8bf*)(ghrow + k0);
        v8bf a1 = *(const v8bf*)(ghrow + k0 + 8);
        v16bf ah = __builtin_shufflevector(a0, a1, 0,1,2,3,4,5,6,7,8,9,10,11,12,13,14,15);
        v8bf l0 = *(const v8bf*)(glrow + k0);
        v8bf l1 = *(const v8bf*)(glrow + k0 + 8);
        v16bf al = __builtin_shufflevector(l0, l1, 0,1,2,3,4,5,6,7,8,9,10,11,12,13,14,15);
        v16bf b0h = *(const v16bf*)(e0h + k0);
        v16bf b0l = *(const v16bf*)(e0l + k0);
        v16bf b1h = *(const v16bf*)(e1h + k0);
        v16bf b1l = *(const v16bf*)(e1l + k0);
        // dot ~= hi*hi + hi*lo + lo*hi (near-fp32), fp32 accumulation
        acc0 = __builtin_amdgcn_wmma_f32_16x16x32_bf16(false, ah, false, b0h, (short)0, acc0, false, false);
        acc1 = __builtin_amdgcn_wmma_f32_16x16x32_bf16(false, ah, false, b1h, (short)0, acc1, false, false);
        acc0 = __builtin_amdgcn_wmma_f32_16x16x32_bf16(false, ah, false, b0l, (short)0, acc0, false, false);
        acc1 = __builtin_amdgcn_wmma_f32_16x16x32_bf16(false, ah, false, b1l, (short)0, acc1, false, false);
        acc0 = __builtin_amdgcn_wmma_f32_16x16x32_bf16(false, al, false, b0h, (short)0, acc0, false, false);
        acc1 = __builtin_amdgcn_wmma_f32_16x16x32_bf16(false, al, false, b1h, (short)0, acc1, false, false);
      }
#pragma unroll
      for (int v = 0; v < 8; ++v) {
        lse_update(rm[v], rs[v], v0 ? (acc0[v] * INV_TEMP) : NEG_FILL);
        lse_update(rm[v], rs[v], v1 ? (acc1[v] * INV_TEMP) : NEG_FILL);
      }
    }

    // merge the 16 per-lane column subsets of each half
#pragma unroll
    for (int v = 0; v < 8; ++v) {
      for (int msk = 1; msk < 16; msk <<= 1) {
        float om = __shfl_xor(rm[v], msk, 32);
        float os = __shfl_xor(rs[v], msk, 32);
        lse_merge(rm[v], rs[v], om, os);
      }
    }
    float lsum = 0.f;
    if (m15 == 0) {
#pragma unroll
      for (int v = 0; v < 8; ++v) {
        if (rvalid[v]) {
          float lse = rm[v] + logf(rs[v]);
          float z = lse - rdiag[v];
          lsum += (z > 0.f) ? (z + log1pf(expf(-z))) : log1pf(expf(z));
        }
      }
    }
    lsum += __shfl_xor(lsum, 16, 32);
    final_v = lsum;
  }
  if (threadIdx.x == 0) part[blk] = final_v;
}

// ---------------------------------------------------------------------------
// Pass 3 (fallback, small workspace): convert-in-loop GEMM. Same math.
// ---------------------------------------------------------------------------
__global__ void __launch_bounds__(32)
gemm_lse_kernel(const float* __restrict__ g, const float* __restrict__ e,
                const float* __restrict__ invg, const float* __restrict__ inve,
                const float* __restrict__ diag,
                const int* __restrict__ pos_list, const int* __restrict__ neg_list,
                const int* __restrict__ npos_a, const int* __restrict__ nneg_a,
                float* __restrict__ part) {
  int blk  = blockIdx.x;
  int b    = blk >> 6;
  int tile = blk & 63;
  int lane = threadIdx.x;
  int m15  = lane & 15;
  int npos = npos_a[b], nneg = nneg_a[b];
  bool active = (nneg > 0) && (tile * 16 < npos);
  float final_v = 0.f;

  if (active) {
    int rslot = tile * 16 + m15;
    int rIdx  = pos_list[b * PDIM + ((rslot < npos) ? rslot : (npos - 1))];
    const float* grow = g + (size_t)(b * PDIM + rIdx) * HDIM;
    int kbaseA = (lane < 16) ? 0 : 16;
    int kbaseB = (lane < 16) ? 0 : 16;
    int half8  = (lane < 16) ? 0 : 8;

    float rowscale[8], rdiag[8]; bool rvalid[8];
#pragma unroll
    for (int v = 0; v < 8; ++v) {
      int slot = tile * 16 + v + half8;
      rvalid[v] = (slot < npos);
      int idx = pos_list[b * PDIM + (rvalid[v] ? slot : (npos - 1))];
      rowscale[v] = invg[b * PDIM + idx];
      rdiag[v]    = diag[b * PDIM + idx];
    }

    float rm[8], rs[8];
#pragma unroll
    for (int v = 0; v < 8; ++v) { rm[v] = -1e30f; rs[v] = 0.f; }

    int nchunk = (nneg + 15) >> 4;
    for (int ch = 0; ch < nchunk; ++ch) {
      int cslot = ch * 16 + m15;
      bool cvalid = (cslot < nneg);
      int cIdx = neg_list[b * PDIM + (cvalid ? cslot : (nneg - 1))];
      const float* erow = e + (size_t)(b * PDIM + cIdx) * HDIM;
      float colscale = inve[b * PDIM + cIdx] * INV_TEMP;

      v8f acc = {};
      for (int k0 = 0; k0 < HDIM; k0 += 32) {
        v16bf ah, al, bh, bl;
        {
          const float* p1 = grow + k0 + kbaseA;
          const float* p2 = p1 + 8;
#pragma unroll
          for (int i = 0; i < 8; ++i) {
            float x = p1[i];
            __bf16 h = f2bf(x);
            ah[i] = h; al[i] = f2bf(x - bf2f(h));
            float y = p2[i];
            __bf16 h2 = f2bf(y);
            ah[i + 8] = h2; al[i + 8] = f2bf(y - bf2f(h2));
          }
        }
        {
          const float* p1 = erow + k0 + kbaseB;
#pragma unroll
          for (int i = 0; i < 16; ++i) {
            float x = p1[i];
            __bf16 h = f2bf(x);
            bh[i] = h; bl[i] = f2bf(x - bf2f(h));
          }
        }
        acc = __builtin_amdgcn_wmma_f32_16x16x32_bf16(false, ah, false, bh, (short)0, acc, false, false);
        acc = __builtin_amdgcn_wmma_f32_16x16x32_bf16(false, ah, false, bl, (short)0, acc, false, false);
        acc = __builtin_amdgcn_wmma_f32_16x16x32_bf16(false, al, false, bh, (short)0, acc, false, false);
      }
#pragma unroll
      for (int v = 0; v < 8; ++v)
        lse_update(rm[v], rs[v], cvalid ? (acc[v] * rowscale[v] * colscale) : NEG_FILL);
    }

#pragma unroll
    for (int v = 0; v < 8; ++v) {
      for (int msk = 1; msk < 16; msk <<= 1) {
        float om = __shfl_xor(rm[v], msk, 32);
        float os = __shfl_xor(rs[v], msk, 32);
        lse_merge(rm[v], rs[v], om, os);
      }
    }
    float lsum = 0.f;
    if (m15 == 0) {
#pragma unroll
      for (int v = 0; v < 8; ++v) {
        if (rvalid[v]) {
          float lse = rm[v] + logf(rs[v]);
          float z = lse - rdiag[v];
          lsum += (z > 0.f) ? (z + log1pf(expf(-z))) : log1pf(expf(z));
        }
      }
    }
    lsum += __shfl_xor(lsum, 16, 32);
    final_v = lsum;
  }
  if (threadIdx.x == 0) part[blk] = final_v;
}

// ---------------------------------------------------------------------------
// Pass 4: deterministic fixed-order reductions + final scalar.
// ---------------------------------------------------------------------------
__global__ void finalize_kernel(const int* __restrict__ labels,
                                const float* __restrict__ ce,
                                const int* __restrict__ npos, const int* __restrict__ nneg,
                                const float* __restrict__ part,
                                float* __restrict__ out) {
  __shared__ float s0[256], s1[256], s2[256];
  int t = threadIdx.x;
  float cs = 0.f, vc = 0.f, ps = 0.f;
  for (int i = t; i < NROW; i += 256) {
    cs += ce[i];
    vc += (labels[i] != IGNORE_L) ? 1.f : 0.f;
  }
  for (int i = t; i < NGEMMBLK; i += 256) ps += part[i];
  s0[t] = cs; s1[t] = vc; s2[t] = ps;
  __syncthreads();
  for (int off = 128; off > 0; off >>= 1) {
    if (t < off) { s0[t] += s0[t + off]; s1[t] += s1[t + off]; s2[t] += s2[t + off]; }
    __syncthreads();
  }
  if (t == 0) {
    float cls = s0[0] / fmaxf(s1[0], 1.f);
    float count = 0.f;
    for (int b = 0; b < BDIM; ++b)
      if (npos[b] > 0 && nneg[b] > 0) count += (float)npos[b];
    float con = (count == 0.f) ? 0.f : (s2[0] / count);
    out[0] = cls + 0.5f * con;
  }
}

// ---------------------------------------------------------------------------
extern "C" void kernel_launch(void* const* d_in, const int* in_sizes, int n_in,
                              void* d_out, int out_size, void* d_ws, size_t ws_size,
                              hipStream_t stream) {
  const float* logits = (const float*)d_in[0];
  const int*   labels = (const int*)d_in[1];
  const float* g      = (const float*)d_in[2];
  const float* e      = (const float*)d_in[3];
  float* out = (float*)d_out;

  // Small workspace region (~1 MB)
  char* ws = (char*)d_ws;
  size_t off = 0;
  float* invg    = (float*)(ws + off); off += (size_t)NROW * 4;
  float* inve    = (float*)(ws + off); off += (size_t)NROW * 4;
  float* diag    = (float*)(ws + off); off += (size_t)NROW * 4;
  float* ce      = (float*)(ws + off); off += (size_t)NROW * 4;
  float* posdiag = (float*)(ws + off); off += (size_t)NROW * 4;
  float* part    = (float*)(ws + off); off += (size_t)NGEMMBLK * 4;
  int* pos_list  = (int*)(ws + off);   off += (size_t)NROW * 4;
  int* neg_list  = (int*)(ws + off);   off += (size_t)NROW * 4;
  int* npos      = (int*)(ws + off);   off += (size_t)BDIM * 4;
  int* nneg      = (int*)(ws + off);   off += (size_t)BDIM * 4;
  off = (off + 255) & ~(size_t)255;

  // Big bf16 region (fast path): 4 matrices of NROW x HDIM bf16 (~192 MB worst case)
  size_t matBytes = (size_t)NROW * HDIM * 2;
  __bf16* Gh = (__bf16*)(ws + off);
  __bf16* Gl = (__bf16*)(ws + off + matBytes);
  __bf16* Eh = (__bf16*)(ws + off + 2 * matBytes);
  __bf16* El = (__bf16*)(ws + off + 3 * matBytes);
  bool fast = (ws_size >= off + 4 * matBytes);

  prep_kernel<<<NROW / 8, 256, 0, stream>>>(logits, labels, g, e, invg, inve, diag, ce);
  compact_kernel<<<BDIM, PDIM, 0, stream>>>(labels, pos_list, neg_list, npos, nneg);
  if (fast) {
    convert_kernel<<<(2 * NROW) / 8, 256, 0, stream>>>(g, e, invg, inve, diag,
                                                       pos_list, neg_list, npos, nneg,
                                                       Gh, Gl, Eh, El, posdiag);
    gemm_lse_fast_kernel<<<NGEMMBLK, 32, 0, stream>>>(Gh, Gl, Eh, El, posdiag,
                                                      npos, nneg, part);
  } else {
    gemm_lse_kernel<<<NGEMMBLK, 32, 0, stream>>>(g, e, invg, inve, diag,
                                                 pos_list, neg_list, npos, nneg, part);
  }
  finalize_kernel<<<1, 256, 0, stream>>>(labels, ce, npos, nneg, part, out);
}
